// QuantLinear_60653528154676
// MI455X (gfx1250) — matrix operand order repaired and checked by reading, not compiled, not run on hardware
//
#include <hip/hip_runtime.h>
#include <stdint.h>

// Problem constants (fixed by the reference)
#define M_DIM 4096
#define K_DIM 4096
#define N_DIM 12288

// Tiling
#define BM 128
#define BN 256
#define BK 64
#define NSTAGE (K_DIM / BK)
#define LDA 72             // f16 stride of ldsA rows (BK + 8 pad)
#define LDB 72             // f16 stride of ldsB rows (BK + 8 pad)

typedef _Float16 half_t;
typedef __attribute__((ext_vector_type(16))) _Float16 v16h;
typedef __attribute__((ext_vector_type(2)))  _Float16 h2;
typedef __attribute__((ext_vector_type(8)))  float    f32x8;

union FragCast { uint4 u[2]; v16h v; };

static __device__ __forceinline__ unsigned int pk_f16(float a, float b) {
#if __has_builtin(__builtin_amdgcn_cvt_pkrtz)
    // v_cvt_pk_rtz_f16_f32; returns __fp16x2 -> take raw bits
    return __builtin_bit_cast(unsigned int, __builtin_amdgcn_cvt_pkrtz(a, b));
#else
    h2 h = { (half_t)a, (half_t)b };
    return __builtin_bit_cast(unsigned int, h);
#endif
}

__global__ __launch_bounds__(256)
void gptq_wmma_f16_kernel(const float* __restrict__ x,
                          const unsigned int* __restrict__ qweight,  // [K/8, N]
                          const unsigned int* __restrict__ qzeros,   // [G, N/8]
                          const float* __restrict__ scales,          // [G, N]
                          const float* __restrict__ bias,            // [N]
                          float* __restrict__ out)                   // [M, N]
{
    __shared__ half_t ldsA[2][BM * LDA];   // x tile, f16, [m][k]
    __shared__ half_t ldsB[2][BN * LDB];   // dequant W tile, f16, [n][k]

    const int tid    = threadIdx.x;
    const int lane   = tid & 31;
    const int wave   = tid >> 5;
    const int wm     = wave & 1;           // 2 waves along M (64 rows each)
    const int wn     = wave >> 1;          // 4 waves along N (64 cols each)
    const int blockM = blockIdx.y * BM;
    const int blockN = blockIdx.x * BN;
    const int waveM  = wm * 64;
    const int waveN  = wn * 64;

    const int acol  = tid & 15;            // float4 column [0,16)
    const int arow0 = tid >> 4;            // base row [0,16)
    const int bn_   = tid;                 // B column within tile [0,256)
    const int gn    = blockN + bn_;        // global output column

    const int laneLo = lane & 15;
    const int laneHi = lane >> 4;

    f32x8 acc[4][4];
    #pragma unroll
    for (int i = 0; i < 4; i++)
        #pragma unroll
        for (int j = 0; j < 4; j++)
            acc[i][j] = (f32x8)0.f;

    // ----- register staging for double buffering -----
    uint2 ah[8];             // x tile, already converted to packed f16
    unsigned int qv[8];      // raw packed int4 weights (one column, 8 rows)
    h2 zz, ss;

    auto stage_load = [&](int k0) {
        #pragma unroll
        for (int i = 0; i < 8; i++) {
            const float4 v = *reinterpret_cast<const float4*>(
                &x[(size_t)(blockM + arow0 + i * 16) * K_DIM + k0 + acol * 4]);
            ah[i] = make_uint2(pk_f16(v.x, v.y), pk_f16(v.z, v.w));
        }
        const int g = k0 >> 7;
        const unsigned int zq = qzeros[(size_t)g * (N_DIM / 8) + (gn >> 3)];
        const float zp1 = (float)(((zq >> ((gn & 7) * 4)) & 0xFu) + 1u);
        const float sc  = scales[(size_t)g * N_DIM + gn];
        const half_t z16 = (half_t)(64.0f + zp1);   // exact integer in f16
        const half_t s16 = (half_t)sc;
        zz = (h2){ z16, z16 };
        ss = (h2){ s16, s16 };
        #pragma unroll
        for (int p = 0; p < 8; p++)
            qv[p] = qweight[(size_t)((k0 >> 3) + p) * N_DIM + gn];
    };

    auto stage_store = [&](int buf) {
        // x f16 into LDS
        #pragma unroll
        for (int i = 0; i < 8; i++)
            *reinterpret_cast<uint2*>(
                &ldsA[buf][(arow0 + i * 16) * LDA + acol * 4]) = ah[i];
        // int4 -> f16 dequant via magic bias 0x5400 (64.0): (64+q/16)-(64+zp1) exact,
        // then one packed multiply by s. v_pk_add_f16 / v_pk_mul_f16, 2 elems/op.
        #pragma unroll
        for (int p = 0; p < 8; p++) {
            const unsigned int q32 = qv[p];
            unsigned int pkw[4];
            #pragma unroll
            for (int pr2 = 0; pr2 < 4; pr2++) {
                const unsigned int qq = q32 >> (8 * pr2);
                const unsigned int t =
                    (qq & 0xFu) | ((qq & 0xF0u) << 12) | 0x54005400u;
                h2 hv = __builtin_bit_cast(h2, t);
                h2 wv = (hv - zz) * ss;
                pkw[pr2] = __builtin_bit_cast(unsigned int, wv);
            }
            *reinterpret_cast<uint4*>(&ldsB[buf][bn_ * LDB + p * 8]) =
                make_uint4(pkw[0], pkw[1], pkw[2], pkw[3]);
        }
    };

    auto compute = [&](int buf) {
        #pragma unroll
        for (int kk = 0; kk < BK; kk += 32) {
            v16h afrag[4];
            #pragma unroll
            for (int mt = 0; mt < 4; mt++) {
                const int m  = waveM + mt * 16 + laneLo;
                const int kb = kk + laneHi * 8;      // lanes16-31: K=8..15 / 24..31
                FragCast fc;
                fc.u[0] = *reinterpret_cast<const uint4*>(&ldsA[buf][m * LDA + kb]);
                fc.u[1] = *reinterpret_cast<const uint4*>(&ldsA[buf][m * LDA + kb + 16]);
                afrag[mt] = fc.v;
            }
            v16h bfrag[4];
            #pragma unroll
            for (int nt = 0; nt < 4; nt++) {
                const int n  = waveN + nt * 16 + laneLo;
                const int kb = kk + laneHi * 8;      // lanes16-31: K=8..15 / 24..31
                FragCast fc;
                fc.u[0] = *reinterpret_cast<const uint4*>(&ldsB[buf][n * LDB + kb]);
                fc.u[1] = *reinterpret_cast<const uint4*>(&ldsB[buf][n * LDB + kb + 16]);
                bfrag[nt] = fc.v;
            }
            #pragma unroll
            for (int nt = 0; nt < 4; nt++)
                #pragma unroll
                for (int mt = 0; mt < 4; mt++)
                    acc[mt][nt] = __builtin_amdgcn_wmma_f32_16x16x32_f16(
                        false, afrag[mt], false, bfrag[nt],
                        (short)0, acc[mt][nt], false, false);
        }
    };

    // ----- double-buffered main loop: one barrier per stage -----
    stage_load(0);
    stage_store(0);
    for (int s = 0; s < NSTAGE; ++s) {
        if (s + 1 < NSTAGE) {
            stage_load((s + 1) * BK);
            if (s + 2 < NSTAGE) {
                __builtin_prefetch(
                    &x[(size_t)(blockM + arow0) * K_DIM + (s + 2) * BK], 0, 1);
                __builtin_prefetch(
                    &qweight[(size_t)(((s + 2) * BK) >> 3) * N_DIM + gn], 0, 1);
            }
        }
        __syncthreads();
        compute(s & 1);
        if (s + 1 < NSTAGE) stage_store((s + 1) & 1);
    }

    // ----- epilogue: bias add + fp32 store -----
    #pragma unroll
    for (int nt = 0; nt < 4; nt++) {
        const int n  = blockN + waveN + nt * 16 + laneLo;
        const float bv = bias[n];
        #pragma unroll
        for (int mt = 0; mt < 4; mt++) {
            #pragma unroll
            for (int r = 0; r < 8; r++) {
                const int m = blockM + waveM + mt * 16 + r + laneHi * 8;
                out[(size_t)m * N_DIM + n] = acc[mt][nt][r] + bv;
            }
        }
    }
}

extern "C" void kernel_launch(void* const* d_in, const int* in_sizes, int n_in,
                              void* d_out, int out_size, void* d_ws, size_t ws_size,
                              hipStream_t stream) {
    const float*        xp = (const float*)d_in[0];
    const unsigned int* qw = (const unsigned int*)d_in[1];
    const unsigned int* qz = (const unsigned int*)d_in[2];
    const float*        sc = (const float*)d_in[3];
    const float*        bi = (const float*)d_in[4];
    float*              op = (float*)d_out;

    dim3 grid(N_DIM / BN, M_DIM / BM);   // 48 x 32 blocks
    gptq_wmma_f16_kernel<<<grid, 256, 0, stream>>>(xp, qw, qz, sc, bi, op);
}
